// Network_86689619903226
// MI455X (gfx1250) — matrix operand order repaired and checked by reading, not compiled, not run on hardware
//
#include <hip/hip_runtime.h>
#include <hip/hip_bf16.h>
#include <stdint.h>

// ---------------------------------------------------------------------------
// Seq2seq GRU (enc 2x128 steps, dec 2x32 steps + vocab logits/CE/argmax)
// MI455X / gfx1250: bf16 WMMA (v_wmma_f32_16x16x32_bf16), wave32.
//   - weights converted f32->bf16 once into d_ws (~116 MB, resident in 192MB L2)
//   - recurrent h kept in f32 (master) + bf16 (WMMA A operand), ping-pong buffers
//   - fused GRU layer kernel: dual GEMM (x@Wih.T, h@Whh.T) + gate math in-regs
// ---------------------------------------------------------------------------

#define BATCH    64
#define HDIM     1024
#define THREEH   3072
#define SIN      128
#define STRG     33
#define VOUT     32000
#define NLAYERS  2

typedef __attribute__((ext_vector_type(16))) __bf16 bf16x16;
typedef __attribute__((ext_vector_type(8)))  __bf16 bf16x8;
typedef __attribute__((ext_vector_type(8)))  float  f32x8;

// float -> bf16 (round-to-nearest-even), stored as raw u16
__device__ __forceinline__ uint16_t f2bf(float f) {
  uint32_t u = __float_as_uint(f);
  u += 0x7FFFu + ((u >> 16) & 1u);
  return (uint16_t)(u >> 16);
}

// ---- WMMA fragment loaders (layouts per CDNA5 ISA §7.12.2) ------
// A 16x32 bf16: lane l: M = row_base + (l&15); lanes 0-15: V0-3 = K0..7,
// V4-7 = K16..23; lanes 16-31 shifted by +8.  Two 16B contiguous loads.
__device__ __forceinline__ bf16x16 load_a_frag(const uint16_t* __restrict__ base,
                                               int ld, int row_base, int k0, int lane) {
  const int m = row_base + (lane & 15);
  const int hg = lane >> 4;
  const __bf16* p = reinterpret_cast<const __bf16*>(base) + (size_t)m * ld + k0 + hg * 8;
  bf16x8 lo = *reinterpret_cast<const bf16x8*>(p);
  bf16x8 hi = *reinterpret_cast<const bf16x8*>(p + 16);
  bf16x16 r;
#pragma unroll
  for (int i = 0; i < 8; ++i) { r[i] = lo[i]; r[i + 8] = hi[i]; }
  return r;
}

// B 32x16 bf16 (K-major weight row): lane l: N = col_base + (l&15);
// lanes 0-15: V0-3 = K0..7, V4-7 = K16..23; lanes 16-31 shifted by +8
// (same K order as A).  Two 16B contiguous loads.
__device__ __forceinline__ bf16x16 load_b_frag(const uint16_t* __restrict__ base,
                                               int ld, int col_base, int k0, int lane) {
  const int n = col_base + (lane & 15);
  const int hg = lane >> 4;
  const __bf16* p = reinterpret_cast<const __bf16*>(base) + (size_t)n * ld + k0 + hg * 8;
  bf16x8 lo = *reinterpret_cast<const bf16x8*>(p);
  bf16x8 hi = *reinterpret_cast<const bf16x8*>(p + 16);
  bf16x16 r;
#pragma unroll
  for (int i = 0; i < 8; ++i) { r[i] = lo[i]; r[i + 8] = hi[i]; }
  return r;
}

__device__ __forceinline__ f32x8 wmma_bf16(bf16x16 a, bf16x16 b, f32x8 c) {
  return __builtin_amdgcn_wmma_f32_16x16x32_bf16(false, a, false, b, (short)0, c,
                                                 false, false);
}

__device__ __forceinline__ float sigmoidf(float x) {
  return 1.0f / (1.0f + __expf(-x));
}

// ---- utility kernels -------------------------------------------------------
__global__ void f32_to_bf16_kernel(const float* __restrict__ src,
                                   uint16_t* __restrict__ dst, int n) {
  int i = blockIdx.x * blockDim.x + threadIdx.x;
  int stride = gridDim.x * blockDim.x;
  for (; i < n; i += stride) dst[i] = f2bf(src[i]);
}

__global__ void zero_h_kernel(float* __restrict__ hf, uint16_t* __restrict__ hb, int n) {
  int i = blockIdx.x * blockDim.x + threadIdx.x;
  int stride = gridDim.x * blockDim.x;
  for (; i < n; i += stride) { hf[i] = 0.0f; hb[i] = 0; }
}

// gather one embedding row per batch element, convert to bf16
__global__ void embed_kernel(const float* __restrict__ table,
                             const int* __restrict__ tokens, int tok_stride,
                             uint16_t* __restrict__ x_bf) {
  const int b = blockIdx.x;
  const int tok = tokens[b * tok_stride];
  const float* src = table + (size_t)tok * HDIM;
  for (int e = threadIdx.x; e < HDIM; e += blockDim.x)
    x_bf[b * HDIM + e] = f2bf(src[e]);
}

// ---- fused GRU layer: gi = x@Wih.T + bih ; gh = h@Whh.T + bhh ; gate math --
// 128 waves: wave = (mg in 0..1 covering 32 batch rows) x (jt in 0..63).
// Each wave: 12 f32x8 accumulators (2 M tiles x r,z,n gates for Wih and Whh), K-loop of 32
// with 12 WMMAs per iteration (B fragments shared across the 2 M tiles).
__global__ __launch_bounds__(256) void gru_layer_kernel(
    const uint16_t* __restrict__ x_bf,    // [64,1024] layer input (bf16)
    const uint16_t* __restrict__ h_bf,    // [64,1024] prev h (bf16)
    const float*    __restrict__ h_f32,   // [64,1024] prev h (f32 master)
    const uint16_t* __restrict__ Wih,     // [3072,1024] bf16 row-major
    const uint16_t* __restrict__ Whh,     // [3072,1024] bf16 row-major
    const float*    __restrict__ bih,     // [3072]
    const float*    __restrict__ bhh,     // [3072]
    uint16_t*       __restrict__ hn_bf,   // [64,1024] new h (bf16)
    float*          __restrict__ hn_f32)  // [64,1024] new h (f32)
{
  const int lane = threadIdx.x & 31;
  const int wid = (blockIdx.x * blockDim.x + threadIdx.x) >> 5;  // 0..127
  const int jt = wid & 63;         // N tile within H
  const int mg = (wid >> 6) & 1;   // covers batch rows [mg*32, mg*32+32)
  const int hg = lane >> 4;
  const int l16 = lane & 15;

  f32x8 acc_i[2][3], acc_h[2][3];
#pragma unroll
  for (int mi = 0; mi < 2; ++mi)
#pragma unroll
    for (int g = 0; g < 3; ++g) { acc_i[mi][g] = (f32x8){}; acc_h[mi][g] = (f32x8){}; }

  for (int k0 = 0; k0 < HDIM; k0 += 32) {
    bf16x16 ax[2], ah[2];
#pragma unroll
    for (int mi = 0; mi < 2; ++mi) {
      const int mb = (mg * 2 + mi) * 16;
      ax[mi] = load_a_frag(x_bf, HDIM, mb, k0, lane);
      ah[mi] = load_a_frag(h_bf, HDIM, mb, k0, lane);
    }
#pragma unroll
    for (int g = 0; g < 3; ++g) {
      bf16x16 bi = load_b_frag(Wih, HDIM, g * HDIM + jt * 16, k0, lane);
#pragma unroll
      for (int mi = 0; mi < 2; ++mi) acc_i[mi][g] = wmma_bf16(ax[mi], bi, acc_i[mi][g]);
      bf16x16 bh = load_b_frag(Whh, HDIM, g * HDIM + jt * 16, k0, lane);
#pragma unroll
      for (int mi = 0; mi < 2; ++mi) acc_h[mi][g] = wmma_bf16(ah[mi], bh, acc_h[mi][g]);
    }
  }

  const int j = jt * 16 + l16;  // column within H (same for all 8 acc rows)
  const float bir = bih[j], biz = bih[HDIM + j], bin = bih[2 * HDIM + j];
  const float bhr = bhh[j], bhz = bhh[HDIM + j], bhn = bhh[2 * HDIM + j];

#pragma unroll
  for (int mi = 0; mi < 2; ++mi) {
    const int mb = (mg * 2 + mi) * 16;
#pragma unroll
    for (int r = 0; r < 8; ++r) {
      const int brow = mb + hg * 8 + r;  // C/D layout: M = 8*(lane>=16) + vgpr
      const float ir = acc_i[mi][0][r] + bir;
      const float iz = acc_i[mi][1][r] + biz;
      const float inn = acc_i[mi][2][r] + bin;
      const float hr = acc_h[mi][0][r] + bhr;
      const float hz = acc_h[mi][1][r] + bhz;
      const float hnn = acc_h[mi][2][r] + bhn;
      const float rg = sigmoidf(ir + hr);
      const float zg = sigmoidf(iz + hz);
      const float ng = tanhf(inn + rg * hnn);
      const float hold = h_f32[brow * HDIM + j];
      const float hv = (1.0f - zg) * ng + zg * hold;
      hn_f32[brow * HDIM + j] = hv;
      hn_bf[brow * HDIM + j] = f2bf(hv);
    }
  }
}

// ---- decoder logits: [64,1024] @ fc[32000,1024].T -> [64,32000] ------------
// 2000 waves, one N tile each; the B (weight) fragment is reused across the
// 4 M tiles (4 WMMAs per K step).
__global__ __launch_bounds__(256) void logits_kernel(
    const uint16_t* __restrict__ h_bf,   // [64,1024]
    const uint16_t* __restrict__ fc_bf,  // [32000,1024]
    float* __restrict__ logits)          // [64,32000]
{
  const int lane = threadIdx.x & 31;
  const int nt = (blockIdx.x * blockDim.x + threadIdx.x) >> 5;  // 0..1999
  const int hg = lane >> 4;
  const int l16 = lane & 15;

  f32x8 acc[4];
#pragma unroll
  for (int m = 0; m < 4; ++m) acc[m] = (f32x8){};

  for (int k0 = 0; k0 < HDIM; k0 += 32) {
    bf16x16 b = load_b_frag(fc_bf, HDIM, nt * 16, k0, lane);
#pragma unroll
    for (int m = 0; m < 4; ++m) {
      bf16x16 a = load_a_frag(h_bf, HDIM, m * 16, k0, lane);
      acc[m] = wmma_bf16(a, b, acc[m]);
    }
  }

  const int col = nt * 16 + l16;
#pragma unroll
  for (int m = 0; m < 4; ++m)
#pragma unroll
    for (int r = 0; r < 8; ++r) {
      const int row = m * 16 + hg * 8 + r;
      logits[(size_t)row * VOUT + col] = acc[m][r];
    }
}

// ---- per-row log-softmax CE + argmax (first-index tie-break) --------------
__global__ __launch_bounds__(256) void softmax_ce_kernel(
    const float* __restrict__ logits, const int* __restrict__ seq_trg, int t,
    float* __restrict__ rowloss, int* __restrict__ top1) {
  const int b = blockIdx.x;
  const int tid = threadIdx.x;
  const float* row = logits + (size_t)b * VOUT;

  float vmax = -3.402823466e+38f;
  int vidx = 0;
  for (int i = tid; i < VOUT; i += 256) {
    float v = row[i];
    if (v > vmax) { vmax = v; vidx = i; }
  }
  __shared__ float smax[256];
  __shared__ int sidx[256];
  smax[tid] = vmax; sidx[tid] = vidx;
  __syncthreads();
  for (int s = 128; s > 0; s >>= 1) {
    if (tid < s) {
      float v2 = smax[tid + s]; int i2 = sidx[tid + s];
      if (v2 > smax[tid] || (v2 == smax[tid] && i2 < sidx[tid])) {
        smax[tid] = v2; sidx[tid] = i2;
      }
    }
    __syncthreads();
  }
  const float m = smax[0];
  const int am = sidx[0];

  float lsum = 0.0f;
  for (int i = tid; i < VOUT; i += 256) lsum += __expf(row[i] - m);
  __shared__ float ssum[256];
  ssum[tid] = lsum;
  __syncthreads();
  for (int s = 128; s > 0; s >>= 1) {
    if (tid < s) ssum[tid] += ssum[tid + s];
    __syncthreads();
  }
  if (tid == 0) {
    const int gold = seq_trg[b * STRG + t + 1];
    const float logp = row[gold] - m - __logf(ssum[0]);
    rowloss[b] = -logp * (1.0f / (float)BATCH);
    top1[b] = am;
  }
}

// loss_t = sum_b rowloss; teacher forcing pick; record top1 in output
__global__ void step_finish_kernel(const float* __restrict__ rowloss,
                                   const int* __restrict__ top1,
                                   const int* __restrict__ seq_trg,
                                   const int* __restrict__ force_teach, int t,
                                   float* __restrict__ losses,
                                   int* __restrict__ trg_cur,
                                   float* __restrict__ d_out) {
  const int tid = threadIdx.x;
  if (tid == 0) {
    float s = 0.0f;
    for (int b = 0; b < BATCH; ++b) s += rowloss[b];
    losses[t] = s;
  }
  if (tid < BATCH) {
    const int gold = seq_trg[tid * STRG + t + 1];
    const int pred = top1[tid];
    trg_cur[tid] = force_teach[t] ? gold : pred;
    d_out[1 + tid * STRG + (t + 1)] = (float)pred;
  }
}

__global__ void dec_init_kernel(const int* __restrict__ seq_trg,
                                int* __restrict__ trg_cur,
                                float* __restrict__ d_out) {
  const int b = threadIdx.x;
  if (b < BATCH) {
    const int v = seq_trg[b * STRG];
    trg_cur[b] = v;
    d_out[1 + b * STRG] = (float)v;
  }
}

__global__ void loss_final_kernel(const float* __restrict__ losses,
                                  float* __restrict__ d_out) {
  if (threadIdx.x == 0) {
    float s = 0.0f;
    for (int i = 0; i < STRG - 1; ++i) s += losses[i];
    d_out[0] = s;
  }
}

// ---------------------------------------------------------------------------
extern "C" void kernel_launch(void* const* d_in, const int* in_sizes, int n_in,
                              void* d_out, int out_size, void* d_ws, size_t ws_size,
                              hipStream_t stream) {
  (void)in_sizes; (void)n_in; (void)out_size; (void)ws_size;

  const int* seq_in = (const int*)d_in[0];
  const int* seq_trg = (const int*)d_in[1];
  const int* force_teach = (const int*)d_in[2];
  const float* emb_in_w = (const float*)d_in[3];
  const float* enc_Wih = (const float*)d_in[4];
  const float* enc_Whh = (const float*)d_in[5];
  const float* enc_bih = (const float*)d_in[6];
  const float* enc_bhh = (const float*)d_in[7];
  const float* emb_out_w = (const float*)d_in[8];
  const float* dec_Wih = (const float*)d_in[9];
  const float* dec_Whh = (const float*)d_in[10];
  const float* dec_bih = (const float*)d_in[11];
  const float* dec_bhh = (const float*)d_in[12];
  const float* fc_w = (const float*)d_in[13];
  float* out = (float*)d_out;

  // ---- workspace carve (requires ~125 MB) ----
  char* p = (char*)d_ws;
  auto carve = [&](size_t bytes) -> char* {
    char* r = p;
    p += (bytes + 255) & ~(size_t)255;
    return r;
  };
  const size_t LW = (size_t)THREEH * HDIM;  // per-layer weight elements
  uint16_t* enc_ih_bf = (uint16_t*)carve(NLAYERS * LW * 2);
  uint16_t* enc_hh_bf = (uint16_t*)carve(NLAYERS * LW * 2);
  uint16_t* dec_ih_bf = (uint16_t*)carve(NLAYERS * LW * 2);
  uint16_t* dec_hh_bf = (uint16_t*)carve(NLAYERS * LW * 2);
  uint16_t* fc_bf     = (uint16_t*)carve((size_t)VOUT * HDIM * 2);
  float*    h_f_buf   = (float*)carve(NLAYERS * 2 * BATCH * HDIM * 4);
  uint16_t* h_b_buf   = (uint16_t*)carve(NLAYERS * 2 * BATCH * HDIM * 2);
  uint16_t* x_bf      = (uint16_t*)carve((size_t)BATCH * HDIM * 2);
  float*    logits    = (float*)carve((size_t)BATCH * VOUT * 4);
  float*    rowloss   = (float*)carve(BATCH * 4);
  int*      top1      = (int*)carve(BATCH * 4);
  int*      trg_cur   = (int*)carve(BATCH * 4);
  float*    losses    = (float*)carve((STRG - 1) * 4);

  float* h_f[NLAYERS][2];
  uint16_t* h_b[NLAYERS][2];
  for (int l = 0; l < NLAYERS; ++l)
    for (int q = 0; q < 2; ++q) {
      h_f[l][q] = h_f_buf + ((size_t)l * 2 + q) * BATCH * HDIM;
      h_b[l][q] = h_b_buf + ((size_t)l * 2 + q) * BATCH * HDIM;
    }

  // ---- one-time (per call) weight conversion + h init ----
  const int nL = (int)(NLAYERS * LW);
  f32_to_bf16_kernel<<<512, 256, 0, stream>>>(enc_Wih, enc_ih_bf, nL);
  f32_to_bf16_kernel<<<512, 256, 0, stream>>>(enc_Whh, enc_hh_bf, nL);
  f32_to_bf16_kernel<<<512, 256, 0, stream>>>(dec_Wih, dec_ih_bf, nL);
  f32_to_bf16_kernel<<<512, 256, 0, stream>>>(dec_Whh, dec_hh_bf, nL);
  f32_to_bf16_kernel<<<1024, 256, 0, stream>>>(fc_w, fc_bf, VOUT * HDIM);
  zero_h_kernel<<<256, 256, 0, stream>>>(h_f_buf, h_b_buf,
                                         NLAYERS * 2 * BATCH * HDIM);

  int pp = 0;

  // ---- encoder: 128 sequential steps x 2 layers ----
  for (int t = 0; t < SIN; ++t) {
    embed_kernel<<<BATCH, 256, 0, stream>>>(emb_in_w, seq_in + t, SIN, x_bf);
    gru_layer_kernel<<<16, 256, 0, stream>>>(
        x_bf, h_b[0][pp], h_f[0][pp], enc_ih_bf, enc_hh_bf, enc_bih, enc_bhh,
        h_b[0][1 - pp], h_f[0][1 - pp]);
    gru_layer_kernel<<<16, 256, 0, stream>>>(
        h_b[0][1 - pp], h_b[1][pp], h_f[1][pp], enc_ih_bf + LW, enc_hh_bf + LW,
        enc_bih + THREEH, enc_bhh + THREEH, h_b[1][1 - pp], h_f[1][1 - pp]);
    pp ^= 1;
  }

  // ---- decoder: 32 sequential steps ----
  dec_init_kernel<<<1, 64, 0, stream>>>(seq_trg, trg_cur, out);
  for (int t = 0; t < STRG - 1; ++t) {
    embed_kernel<<<BATCH, 256, 0, stream>>>(emb_out_w, trg_cur, 1, x_bf);
    gru_layer_kernel<<<16, 256, 0, stream>>>(
        x_bf, h_b[0][pp], h_f[0][pp], dec_ih_bf, dec_hh_bf, dec_bih, dec_bhh,
        h_b[0][1 - pp], h_f[0][1 - pp]);
    gru_layer_kernel<<<16, 256, 0, stream>>>(
        h_b[0][1 - pp], h_b[1][pp], h_f[1][pp], dec_ih_bf + LW, dec_hh_bf + LW,
        dec_bih + THREEH, dec_bhh + THREEH, h_b[1][1 - pp], h_f[1][1 - pp]);
    logits_kernel<<<250, 256, 0, stream>>>(h_b[1][1 - pp], fc_bf, logits);
    softmax_ce_kernel<<<BATCH, 256, 0, stream>>>(logits, seq_trg, t, rowloss, top1);
    step_finish_kernel<<<1, 64, 0, stream>>>(rowloss, top1, seq_trg, force_teach,
                                             t, losses, trg_cur, out);
    pp ^= 1;
  }

  loss_final_kernel<<<1, 32, 0, stream>>>(losses, out);
}
